// EncoderBiRNN_60370060312742
// MI455X (gfx1250) — compile-verified, matrix operand order checked by reading, not run on hardware
//
#include <hip/hip_runtime.h>
#include <hip/hip_bf16.h>
#include <math.h>

// ---------------- types ----------------
typedef __attribute__((ext_vector_type(16))) __bf16 v16bf;
typedef __attribute__((ext_vector_type(8)))  float  v8f;

union ABu { v16bf v; uint4 q[2]; };

__device__ __forceinline__ unsigned short f2bf(float f) {
  unsigned u = __float_as_uint(f);
  u += 0x7FFFu + ((u >> 16) & 1u);   // round-to-nearest-even
  return (unsigned short)(u >> 16);
}

// Async global->LDS copy (CDNA5 TDM-adjacent path, ISA §15.18 / §4 async).
// vdst VGPR = wave-relative LDS byte address (low 32 bits of generic LDS ptr),
// vaddr = 64-bit global address, saddr = off. Tracked by ASYNCcnt.
__device__ __forceinline__ void async_ld_b128(const void* g, void* lds) {
  asm volatile("global_load_async_to_lds_b128 %0, %1, off"
               :
               : "v"((unsigned)(size_t)lds),
                 "v"((unsigned long long)(size_t)g)
               : "memory");
}
__device__ __forceinline__ void async_wait0() {
  asm volatile("s_wait_asynccnt 0" ::: "memory");
}

// ---------------- problem constants ----------------
#define NB    32
#define LSEQ  512
#define HDIM  512
#define EDIM  1024
#define G3    1536
#define ROWS  (LSEQ * NB)          // 16384
#define OUT_HID_OFF 16777216ull    // floats: outputs (N*L*1024) before hiddens

// ws byte offsets
#define OFF_XBF  0ull              // 16384*1024 bf16 = 32 MB
#define OFF_WIH  33554432ull       // 2*2*1536*1024 bf16
#define OFF_WHH  46137344ull       // 2*2*1536*512 bf16
#define OFF_HCUR 52428800ull       // 2*32*512 bf16 h-carry
#define OFF_CNT  52494336ull       // barrier counter
#define OFF_GI   52494592ull       // 2*16384*1536 f32 = 201 MB

// ---------------- embedding gather -> bf16, layout [t][n][e] ----------------
__global__ __launch_bounds__(256) void gather_emb_k(
    const int* __restrict__ x, const float* __restrict__ emb,
    unsigned short* __restrict__ Xbf) {
  size_t i = (size_t)blockIdx.x * 256 + threadIdx.x;  // exactly 16384*1024
  int row = (int)(i >> 10);
  int e   = (int)(i & 1023);
  int t = row >> 5, n = row & 31;
  int tok = x[n * LSEQ + t];
  Xbf[i] = f2bf(emb[(size_t)tok * EDIM + e]);
}

// ---------------- weight f32 -> bf16 ----------------
__global__ __launch_bounds__(256) void cvt_weights_k(
    const float* __restrict__ wih, const float* __restrict__ whh,
    unsigned short* __restrict__ wihb, unsigned short* __restrict__ whhb) {
  size_t i = (size_t)blockIdx.x * 256 + threadIdx.x;
  const size_t NI = 6291456, NH = 3145728;
  if (i < NI) wihb[i] = f2bf(wih[i]);
  else { size_t j = i - NI; if (j < NH) whhb[j] = f2bf(whh[j]); }
}

// ---------------- init h-carry + barrier counter ----------------
__global__ __launch_bounds__(256) void init_rec_k(unsigned short* hcur,
                                                  unsigned int* cnt) {
  int i = blockIdx.x * 256 + threadIdx.x;
  if (i == 0) *cnt = 0u;
  if (i < 2 * NB * HDIM) hcur[i] = (unsigned short)0;
}

// ---------------- big input-side GEMM: Gi[d][t*N+n][g] = A·Wihᵀ + b_ih --------
// A rows = L*N = 16384, cols = 1024; Wih = (1536 x 1024); block tile 128x128.
template <int LAYER>
__global__ __launch_bounds__(256) void gi_gemm_k(
    const unsigned short* __restrict__ Xbf,   // layer0 input (bf16)
    const float* __restrict__ hid,            // layer0 hiddens (f32) for LAYER=1
    const unsigned short* __restrict__ wihb,
    const float* __restrict__ bih,
    float* __restrict__ Gi) {
  __shared__ uint4 AtQ[512];   // 128 x 32 bf16
  __shared__ uint4 BtQ[512];   // 128 gates x 32 k (bf16)
  unsigned short* At = (unsigned short*)AtQ;
  unsigned short* Bt = (unsigned short*)BtQ;

  const int bm = blockIdx.x, bn = blockIdx.y, d = blockIdx.z;
  const int tid = threadIdx.x, lane = tid & 31, wave = tid >> 5;
  const int wm = wave >> 2, wn = wave & 3;       // 2 x 4 wave grid
  const int m = lane & 15, hlf = lane >> 4;
  const unsigned short* Wb =
      wihb + ((size_t)(LAYER * 2 + d) * G3 + (size_t)bn * 128) * EDIM;

  const v8f vzero = {0.f, 0.f, 0.f, 0.f, 0.f, 0.f, 0.f, 0.f};
  v8f acc[4][2];
  for (int i2 = 0; i2 < 4; ++i2)
    for (int j2 = 0; j2 < 2; ++j2) acc[i2][j2] = vzero;

  for (int kc = 0; kc < 32; ++kc) {
    if (LAYER == 0) {
      for (int c = tid; c < 512; c += 256) {
        int r = c >> 2, q = c & 3;
        async_ld_b128(Xbf + (size_t)(bm * 128 + r) * EDIM + kc * 32 + q * 8,
                      &AtQ[r * 4 + q]);
      }
    } else {
      int dir0 = (kc * 32) >> 9;   // layer-0 direction this k-chunk reads
      int e0 = (kc * 32) & 511;
      for (int c = tid; c < 4096; c += 256) {
        int r = c >> 5, j = c & 31;
        int grow = bm * 128 + r;
        int t = grow >> 5, n = grow & 31;
        float v = hid[(((size_t)t * 4 + dir0) * NB + n) * HDIM + e0 + j];
        At[r * 32 + j] = f2bf(v);
      }
    }
    for (int c = tid; c < 512; c += 256) {
      int g = c >> 2, q = c & 3;
      async_ld_b128(Wb + (size_t)g * EDIM + kc * 32 + q * 8, &BtQ[g * 4 + q]);
    }
    async_wait0();
    __syncthreads();

    ABu a[4], b[2];
#pragma unroll
    for (int i2 = 0; i2 < 4; ++i2) {
      const unsigned short* ar = At + (size_t)(wm * 64 + i2 * 16 + m) * 32;
      a[i2].q[0] = *(const uint4*)(ar + 8 * hlf);
      a[i2].q[1] = *(const uint4*)(ar + 16 + 8 * hlf);
    }
#pragma unroll
    for (int j2 = 0; j2 < 2; ++j2) {
      const unsigned short* br = Bt + (size_t)(wn * 32 + j2 * 16 + m) * 32;
      b[j2].q[0] = *(const uint4*)(br + 8 * hlf);
      b[j2].q[1] = *(const uint4*)(br + 16 + 8 * hlf);
    }
#pragma unroll
    for (int i2 = 0; i2 < 4; ++i2)
#pragma unroll
      for (int j2 = 0; j2 < 2; ++j2)
        acc[i2][j2] = __builtin_amdgcn_wmma_f32_16x16x32_bf16(
            false, a[i2].v, false, b[j2].v, (short)0, acc[i2][j2], false, false);
    __syncthreads();
  }

  const float* bih_d = bih + (size_t)(LAYER * 2 + d) * G3;
  float* GiD = Gi + (size_t)d * ROWS * G3;
#pragma unroll
  for (int i2 = 0; i2 < 4; ++i2)
#pragma unroll
    for (int j2 = 0; j2 < 2; ++j2) {
      int col = bn * 128 + wn * 32 + j2 * 16 + m;
      float bb = bih_d[col];
#pragma unroll
      for (int q = 0; q < 8; ++q) {
        int row = bm * 128 + wm * 64 + i2 * 16 + q + 8 * hlf;
        GiD[(size_t)row * G3 + col] = acc[i2][j2][q] + bb;
      }
    }
}

// ---------------- persistent recurrence kernel (one layer, both dirs) --------
// 32 WGs: dir = wg/16, chunk = wg%16 -> h columns [chunk*32, chunk*32+32)
// per step: gh slice (32 x 96 gates) = h_t(32x512) · Whhᵀ via WMMA bf16.
__global__ __launch_bounds__(128) void gru_rec_k(
    const unsigned short* __restrict__ whhb, const float* __restrict__ bhh,
    const float* __restrict__ Gi, unsigned short* __restrict__ hcur,
    unsigned int* __restrict__ cnt, float* __restrict__ dout, int layer) {
  __shared__ uint4 HtQ[2048];       // 32 x 512 bf16 (h_t)
  __shared__ float Gh[NB * 96];     // gh slice
  unsigned short* Ht = (unsigned short*)HtQ;

  const int wg = blockIdx.x;
  const int dir = wg >> 4;
  const int chunk = wg & 15;
  const int tid = threadIdx.x;
  const int lane = tid & 31;
  const int wave = tid >> 5;        // 4 waves, 3 output tiles each
  const int m = lane & 15;
  const int hlf = lane >> 4;
  const int li = layer * 2 + dir;

  // constant per-lane B (w_hh) row base pointers for this wave's 3 tiles
  const unsigned short* bptr[3];
#pragma unroll
  for (int s = 0; s < 3; ++s) {
    int tt = wave * 3 + s;
    int tn = tt >> 1;
    int gl = tn * 16 + m;                               // 0..95 local gate
    int g = (gl >> 5) * HDIM + chunk * 32 + (gl & 31);  // r/z/n blocks
    bptr[s] = whhb + ((size_t)li * G3 + g) * HDIM;
  }
  const float* bhh_d = bhh + (size_t)li * G3;
  float* hid = dout + OUT_HID_OFF;
  unsigned short* hc_d = hcur + (size_t)dir * NB * HDIM;

  for (int t = 0; t < LSEQ; ++t) {
    // ---- stage h_t into LDS via async global->LDS ----
    for (int idx = tid; idx < 2048; idx += 128)
      async_ld_b128(hc_d + (size_t)idx * 8, &HtQ[idx]);
    async_wait0();
    __syncthreads();

    // ---- gh = h_t · Whhᵀ (WMMA bf16, K=512 in 16 steps) ----
#pragma unroll
    for (int s = 0; s < 3; ++s) {
      int tt = wave * 3 + s;
      int tm = tt & 1;
      int tn = tt >> 1;
      const unsigned short* arow = Ht + (size_t)(tm * 16 + m) * HDIM;
      const unsigned short* brow = bptr[s];
      ABu a, b;
      v8f acc = {0.f, 0.f, 0.f, 0.f, 0.f, 0.f, 0.f, 0.f};
#pragma unroll
      for (int kc = 0; kc < 16; ++kc) {
        a.q[0] = *(const uint4*)(arow + kc * 32 + 8 * hlf);
        a.q[1] = *(const uint4*)(arow + kc * 32 + 16 + 8 * hlf);
        b.q[0] = *(const uint4*)(brow + kc * 32 + 8 * hlf);
        b.q[1] = *(const uint4*)(brow + kc * 32 + 16 + 8 * hlf);
        acc = __builtin_amdgcn_wmma_f32_16x16x32_bf16(false, a.v, false, b.v,
                                                      (short)0, acc, false, false);
      }
#pragma unroll
      for (int j = 0; j < 8; ++j) {
        int row = tm * 16 + j + 8 * hlf;
        int col = tn * 16 + m;
        Gh[row * 96 + col] = acc[j];
      }
    }
    __syncthreads();

    // ---- elementwise GRU update for our 32 h-columns ----
    const float* gi = Gi + ((size_t)dir * ROWS + (size_t)t * NB) * G3;
    for (int e = tid; e < NB * 32; e += 128) {
      int n = e >> 5;
      int hcc = e & 31;
      int hg = chunk * 32 + hcc;
      float gr = gi[(size_t)n * G3 + hg] + Gh[n * 96 + hcc] + bhh_d[hg];
      float gz = gi[(size_t)n * G3 + 512 + hg] + Gh[n * 96 + 32 + hcc] + bhh_d[512 + hg];
      float gn = gi[(size_t)n * G3 + 1024 + hg];
      float ghn = Gh[n * 96 + 64 + hcc] + bhh_d[1024 + hg];
      float r = 1.0f / (1.0f + __expf(-gr));
      float z = 1.0f / (1.0f + __expf(-gz));
      float nn = tanhf(gn + r * ghn);
      float hprev = (t == 0) ? 0.0f
                             : hid[(((size_t)(t - 1) * 4 + li) * NB + n) * HDIM + hg];
      float hnew = (1.0f - z) * nn + z * hprev;
      hid[(((size_t)t * 4 + li) * NB + n) * HDIM + hg] = hnew;
      hc_d[(size_t)n * HDIM + hg] = f2bf(hnew);
      if (layer == 1)
        dout[(size_t)n * (LSEQ * 1024) + (size_t)t * 1024 + dir * HDIM + hg] = hnew;
    }

    // ---- grid-wide barrier across the 32 workgroups ----
    __threadfence();
    __syncthreads();
    if (tid == 0) {
      atomicAdd(cnt, 1u);
      unsigned tgt = (unsigned)(t + 1) * 32u;
      while (__hip_atomic_load(cnt, __ATOMIC_ACQUIRE, __HIP_MEMORY_SCOPE_AGENT) < tgt)
        __builtin_amdgcn_s_sleep(2);
    }
    __syncthreads();
  }
}

// ---------------- launcher ----------------
extern "C" void kernel_launch(void* const* d_in, const int* in_sizes, int n_in,
                              void* d_out, int out_size, void* d_ws, size_t ws_size,
                              hipStream_t stream) {
  (void)in_sizes; (void)n_in; (void)out_size; (void)ws_size;
  const int* x = (const int*)d_in[0];
  const float* emb = (const float*)d_in[1];
  const float* wih = (const float*)d_in[2];
  const float* whh = (const float*)d_in[3];
  const float* bih = (const float*)d_in[4];
  const float* bhh = (const float*)d_in[5];
  float* out = (float*)d_out;
  char* ws = (char*)d_ws;

  unsigned short* Xbf  = (unsigned short*)(ws + OFF_XBF);
  unsigned short* wihb = (unsigned short*)(ws + OFF_WIH);
  unsigned short* whhb = (unsigned short*)(ws + OFF_WHH);
  unsigned short* hcur = (unsigned short*)(ws + OFF_HCUR);
  unsigned int*   cnt  = (unsigned int*)(ws + OFF_CNT);
  float*          Gi   = (float*)(ws + OFF_GI);
  float*          hid  = out + OUT_HID_OFF;

  gather_emb_k<<<65536, 256, 0, stream>>>(x, emb, Xbf);
  cvt_weights_k<<<36864, 256, 0, stream>>>(wih, whh, wihb, whhb);

  // layer 0
  gi_gemm_k<0><<<dim3(128, 12, 2), 256, 0, stream>>>(Xbf, hid, wihb, bih, Gi);
  init_rec_k<<<128, 256, 0, stream>>>(hcur, cnt);
  gru_rec_k<<<32, 128, 0, stream>>>(whhb, bhh, Gi, hcur, cnt, out, 0);

  // layer 1 (input = layer-0 hiddens, read from d_out)
  gi_gemm_k<1><<<dim3(128, 12, 2), 256, 0, stream>>>(Xbf, hid, wihb, bih, Gi);
  init_rec_k<<<128, 256, 0, stream>>>(hcur, cnt);
  gru_rec_k<<<32, 128, 0, stream>>>(whhb, bhh, Gi, hcur, cnt, out, 1);
}
